// MultiHeadedAttention_2989297238062
// MI455X (gfx1250) — matrix operand order repaired and checked by reading, not compiled, not run on hardware
//
#include <hip/hip_runtime.h>

// ---------------------------------------------------------------------------
// MultiHeadedAttention forward for gfx1250 (MI455X), wave32 + WMMA f16 + TDM
// B=16, S=1024, D=512, H=8, DK=64
// ---------------------------------------------------------------------------

#define B_SZ   16
#define S_LEN  1024
#define D_MOD  512
#define NHEAD  8
#define DK     64

typedef __attribute__((ext_vector_type(16))) _Float16     v16h;
typedef __attribute__((ext_vector_type(8)))  _Float16     v8h;
typedef __attribute__((ext_vector_type(8)))  float        v8f;
typedef __attribute__((ext_vector_type(4)))  unsigned int u32x4;
typedef __attribute__((ext_vector_type(8)))  int          i32x8;
typedef __attribute__((ext_vector_type(4)))  int          i32x4;

__device__ __forceinline__ v8f wmma_f16(v16h a, v16h b, v8f c) {
    return __builtin_amdgcn_wmma_f32_16x16x32_f16(
        false, a, false, b, (short)0, c, false, false);
}

__device__ __forceinline__ v16h make_a_f16(const _Float16* p0, const _Float16* p1) {
    v8h lo = *(const v8h*)p0;
    v8h hi = *(const v8h*)p1;
    v16h a;
#pragma unroll
    for (int i = 0; i < 8; ++i) { a[i] = lo[i]; a[i + 8] = hi[i]; }
    return a;
}

__device__ __forceinline__ v16h make_a_f32(const float* p0, const float* p1) {
    v16h a;
#pragma unroll
    for (int i = 0; i < 8; ++i) {
        a[i]     = (_Float16)p0[i];
        a[i + 8] = (_Float16)p1[i];
    }
    return a;
}

// ---------------------------------------------------------------------------
// Tensor Data Mover: 2D tile (f16 elements) global -> LDS.
// D# per CDNA5 ISA §8.3/8.4: group0 {flags, lds_addr, global_addr, type=2},
// group1 {data_size=2B, tensor_dim0/1, tile_dim0/1, tensor_dim0_stride}.
// This toolchain exposes the 6-arg builtin:
//   (u32x4 g0, i32x8 g1, i32x4 g2, i32x4 g3, i32x8 g4, i32 cpol)
// ---------------------------------------------------------------------------
__device__ __forceinline__ void tdm_load_2d_f16(
    unsigned lds_off, const void* gaddr,
    unsigned tensor_d0, unsigned tensor_d1,
    unsigned tile_d0, unsigned tile_d1,
    unsigned long long stride0_elems) {
    unsigned long long ga = (unsigned long long)gaddr;
    u32x4 g0;
    g0[0] = 1u;                                            // count=1, user D#
    g0[1] = lds_off;                                       // LDS byte address
    g0[2] = (unsigned)(ga & 0xFFFFFFFFu);                  // global_addr[31:0]
    g0[3] = (unsigned)((ga >> 32) & 0x1FFFFFFu) | (2u << 30); // addr[56:32], type=2
    i32x8 g1;
    g1[0] = (int)(1u << 16);                               // data_size=1 -> 2 bytes
    g1[1] = (int)((tensor_d0 & 0xFFFFu) << 16);            // tensor_dim0[15:0]
    g1[2] = (int)(((tensor_d0 >> 16) & 0xFFFFu) |
                  ((tensor_d1 & 0xFFFFu) << 16));          // dim0[31:16] | dim1[15:0]
    g1[3] = (int)(((tensor_d1 >> 16) & 0xFFFFu) |
                  ((tile_d0 & 0xFFFFu) << 16));            // dim1[31:16] | tile_dim0
    g1[4] = (int)(tile_d1 & 0xFFFFu);                      // tile_dim1, tile_dim2=0
    g1[5] = (int)(unsigned)(stride0_elems & 0xFFFFFFFFull);// dim0_stride[31:0]
    g1[6] = (int)(unsigned)((stride0_elems >> 32) & 0xFFFFull); // dim0_stride[47:32]
    g1[7] = 0;
    i32x4 gz4 = {0, 0, 0, 0};                              // 2D: groups 2/3 unused
    i32x8 gz8 = {0, 0, 0, 0, 0, 0, 0, 0};
    __builtin_amdgcn_tensor_load_to_lds(g0, g1, gz4, gz4, gz8, 0);
}

__device__ __forceinline__ unsigned lds_offset_of(const void* p) {
    // Flat shared-aperture address: addr[31:0] is the LDS byte offset (ISA 10.2).
    return (unsigned)(unsigned long long)(size_t)p;
}

// ---------------------------------------------------------------------------
// Kernel 0: convert the four 512x512 fp32 weight matrices to f16
// ---------------------------------------------------------------------------
__global__ __launch_bounds__(256) void cvt_weights(
    const float* __restrict__ Wq, const float* __restrict__ Wk,
    const float* __restrict__ Wv, const float* __restrict__ Wo,
    _Float16* __restrict__ W16) {
    int i = blockIdx.x * 256 + threadIdx.x;
    W16[i]              = (_Float16)Wq[i];
    W16[262144 + i]     = (_Float16)Wk[i];
    W16[2 * 262144 + i] = (_Float16)Wv[i];
    W16[3 * 262144 + i] = (_Float16)Wo[i];
}

// ---------------------------------------------------------------------------
// Kernel 1: projection  dst = X @ W^T + b ; one wave = 32x64 output tile
// (2 A-frags x 4 B-frags -> 8 WMMAs per 32-wide K step)
//   mode 0: dst f16 [B,H,S,DK] (Q,K);  mode 1: dst f16 [B,H,DK,S] (V^T)
// ---------------------------------------------------------------------------
__global__ __launch_bounds__(256) void proj_kernel(
    const float* __restrict__ X, const _Float16* __restrict__ W16,
    const float* __restrict__ bias, _Float16* __restrict__ dst, int mode) {
    const int wid  = blockIdx.x * 8 + threadIdx.y;     // 0 .. 4095
    const int tm   = wid >> 3;                         // 0 .. 511  (32-row tile)
    const int tn   = wid & 7;                          // 0 .. 7    (64-col tile)
    const int lane = threadIdx.x;
    const int l15  = lane & 15;
    const int hi8  = (lane & 16) ? 8 : 0;
    const int hi16 = hi8 << 1;

    const float*    xrow0 = X + (size_t)(tm * 32 + l15) * D_MOD;
    const float*    xrow1 = xrow0 + (size_t)16 * D_MOD;
    const int       cn    = tn * 64;
    const _Float16* wrow  = W16 + (size_t)(cn + l15) * D_MOD + hi8;

    v8f acc[2][4];
#pragma unroll
    for (int i = 0; i < 2; ++i)
#pragma unroll
        for (int n = 0; n < 4; ++n) acc[i][n] = (v8f){};

    for (int ks = 0; ks < D_MOD; ks += 32) {
        if (ks + 32 < D_MOD) __builtin_prefetch(xrow0 + ks + 32, 0, 1);
        v16h a0 = make_a_f32(xrow0 + ks + hi8, xrow0 + ks + 16 + hi8);
        v16h a1 = make_a_f32(xrow1 + ks + hi8, xrow1 + ks + 16 + hi8);
#pragma unroll
        for (int n = 0; n < 4; ++n) {
            v16h b = make_a_f16(wrow + (size_t)n * 16 * D_MOD + ks, wrow + (size_t)n * 16 * D_MOD + ks + 16);
            acc[0][n] = wmma_f16(a0, b, acc[0][n]);
            acc[1][n] = wmma_f16(a1, b, acc[1][n]);
        }
    }

    const int head = tn;                               // 64-col block == one head
#pragma unroll
    for (int n = 0; n < 4; ++n) {
        const int   j  = cn + n * 16 + l15;
        const float bj = bias[j];
        const int   d  = n * 16 + l15;
#pragma unroll
        for (int i = 0; i < 2; ++i) {
#pragma unroll
            for (int r = 0; r < 8; ++r) {
                const int nr = tm * 32 + i * 16 + r + hi8;
                const int bb = nr >> 10;
                const int ss = nr & 1023;
                const float val = acc[i][n][r] + bj;
                if (mode == 0)
                    dst[(((size_t)(bb * NHEAD + head)) * S_LEN + ss) * DK + d] =
                        (_Float16)val;
                else
                    dst[(((size_t)(bb * NHEAD + head)) * DK + d) * S_LEN + ss] =
                        (_Float16)val;
            }
        }
    }
}

// ---------------------------------------------------------------------------
// Kernel 2: flash attention. Block = 8 waves sharing one (b,h); each wave owns
// a 16-row q tile. K (32x64) and V^T (64x32) tiles are TDM-staged into LDS,
// double buffered, shared by all 8 waves.
// ---------------------------------------------------------------------------
__global__ __launch_bounds__(256) void attn_kernel(
    const _Float16* __restrict__ q16, const _Float16* __restrict__ k16,
    const _Float16* __restrict__ vT16, _Float16* __restrict__ att16) {
    __shared__ _Float16 Kbuf[2][32 * 64];              // 8 KB  (rows s, cols d)
    __shared__ _Float16 Vbuf[2][64 * 32];              // 8 KB  (rows d, cols s)
    __shared__ _Float16 Pbuf[8][16 * 32];              // 8 KB  per-wave P staging

    const int bh   = blockIdx.x >> 3;                  // 0 .. 127
    const int qg   = blockIdx.x & 7;                   // q-tile group
    const int ty   = threadIdx.y;
    const int qt   = qg * 8 + ty;                      // this wave's q tile
    const int lane = threadIdx.x;
    const int l15  = lane & 15;
    const int hi8  = (lane & 16) ? 8 : 0;
    const int hi16 = hi8 << 1;

    const _Float16* kbase = k16  + (size_t)bh * S_LEN * DK;
    const _Float16* vbase = vT16 + (size_t)bh * DK * S_LEN;

    // Q A-fragments (loaded once)
    const _Float16* qrow = q16 + (size_t)(bh * S_LEN + qt * 16 + l15) * DK;
    const v16h qa0 = make_a_f16(qrow + hi8,      qrow + 16 + hi8);
    const v16h qa1 = make_a_f16(qrow + 32 + hi8, qrow + 48 + hi8);

    float m[8], l[8];
    v8f acc[4];
#pragma unroll
    for (int r = 0; r < 8; ++r) { m[r] = -1e30f; l[r] = 0.0f; }
#pragma unroll
    for (int dt = 0; dt < 4; ++dt) acc[dt] = (v8f){};

    const int NIT = S_LEN / 32;                        // 32 iterations

    // prologue: stage tile 0 into buffer 0 (one wave drives the TDM)
    if (ty == 0) {
        tdm_load_2d_f16(lds_offset_of(&Kbuf[0][0]), kbase,
                        DK, S_LEN, DK, 32, DK);
        tdm_load_2d_f16(lds_offset_of(&Vbuf[0][0]), vbase,
                        S_LEN, DK, 32, DK, S_LEN);
    }

    for (int it = 0; it < NIT; ++it) {
        const int cur = it & 1;
        if (ty == 0) {
            if (it + 1 < NIT) {
                const int kc2 = (it + 1) * 32;
                tdm_load_2d_f16(lds_offset_of(&Kbuf[cur ^ 1][0]),
                                kbase + (size_t)kc2 * DK, DK, S_LEN, DK, 32, DK);
                tdm_load_2d_f16(lds_offset_of(&Vbuf[cur ^ 1][0]),
                                vbase + kc2, S_LEN, DK, 32, DK, S_LEN);
                __builtin_amdgcn_s_wait_tensorcnt(2);  // current pair done
            } else {
                __builtin_amdgcn_s_wait_tensorcnt(0);
            }
        }
        __syncthreads();                               // buffer `cur` visible

        // ---- scores: 16 q-rows x 32 key-cols from Kbuf ---------------------
        const _Float16* kp0 = &Kbuf[cur][(size_t)l15 * DK + hi8];
        const _Float16* kp1 = &Kbuf[cur][(size_t)(16 + l15) * DK + hi8];
        v8f s0 = {}, s1 = {};
        s0 = wmma_f16(qa0, make_a_f16(kp0, kp0 + 16),      s0);
        s0 = wmma_f16(qa1, make_a_f16(kp0 + 32, kp0 + 48), s0);
        s1 = wmma_f16(qa0, make_a_f16(kp1, kp1 + 16),      s1);
        s1 = wmma_f16(qa1, make_a_f16(kp1 + 32, kp1 + 48), s1);

        // ---- online softmax (rows live in 16-lane halves) ------------------
#pragma unroll
        for (int r = 0; r < 8; ++r) {
            float a = s0[r] * 0.125f;                  // 1/sqrt(64)
            float b = s1[r] * 0.125f;
            float mt = fmaxf(a, b);
#pragma unroll
            for (int off = 1; off < 16; off <<= 1)
                mt = fmaxf(mt, __shfl_xor(mt, off, 16));
            const float mn = fmaxf(m[r], mt);
            const float f  = __expf(m[r] - mn);
            m[r] = mn;
            const float p0 = __expf(a - mn);
            const float p1 = __expf(b - mn);
            float rs = p0 + p1;
#pragma unroll
            for (int off = 1; off < 16; off <<= 1)
                rs += __shfl_xor(rs, off, 16);
            l[r] = l[r] * f + rs;
#pragma unroll
            for (int dt = 0; dt < 4; ++dt) acc[dt][r] *= f;
            const int M = r + hi8;
            Pbuf[ty][M * 32 + l15]      = (_Float16)p0;
            Pbuf[ty][M * 32 + 16 + l15] = (_Float16)p1;
        }

        // reload P as 16x32 A-fragment (wave-private LDS region)
        const _Float16* pl = &Pbuf[ty][l15 * 32 + hi8];
        const v16h pa = make_a_f16(pl, pl + 16);

        // ---- PV: acc += P(16x32) @ V(32x64) from Vbuf ----------------------
#pragma unroll
        for (int dt = 0; dt < 4; ++dt) {
            const _Float16* vp = &Vbuf[cur][(size_t)(dt * 16 + l15) * 32 + hi8];
            acc[dt] = wmma_f16(pa, make_a_f16(vp, vp + 16), acc[dt]);
        }
        __syncthreads();                               // done with buffer `cur`
    }

    // ---- normalize + store as [B, S, H*DK] f16 -----------------------------
    const int bb = bh >> 3;
    const int hh = bh & 7;
#pragma unroll
    for (int r = 0; r < 8; ++r) {
        const float inv = 1.0f / l[r];
        const int ss = qt * 16 + r + hi8;
        _Float16* op = att16 + ((size_t)(bb * S_LEN + ss)) * D_MOD + hh * DK + l15;
#pragma unroll
        for (int dt = 0; dt < 4; ++dt)
            op[dt * 16] = (_Float16)(acc[dt][r] * inv);
    }
}

// ---------------------------------------------------------------------------
// Kernel 3: output projection  out = att16 @ Wo^T + bo (fp32), 32x64 per wave
// ---------------------------------------------------------------------------
__global__ __launch_bounds__(256) void oproj_kernel(
    const _Float16* __restrict__ att16, const _Float16* __restrict__ Wo16,
    const float* __restrict__ bo, float* __restrict__ out) {
    const int wid  = blockIdx.x * 8 + threadIdx.y;
    const int tm   = wid >> 3;
    const int tn   = wid & 7;
    const int lane = threadIdx.x;
    const int l15  = lane & 15;
    const int hi8  = (lane & 16) ? 8 : 0;
    const int hi16 = hi8 << 1;

    const _Float16* arow0 = att16 + (size_t)(tm * 32 + l15) * D_MOD;
    const _Float16* arow1 = arow0 + (size_t)16 * D_MOD;
    const int       cn    = tn * 64;
    const _Float16* wrow  = Wo16 + (size_t)(cn + l15) * D_MOD + hi8;

    v8f acc[2][4];
#pragma unroll
    for (int i = 0; i < 2; ++i)
#pragma unroll
        for (int n = 0; n < 4; ++n) acc[i][n] = (v8f){};

    for (int ks = 0; ks < D_MOD; ks += 32) {
        v16h a0 = make_a_f16(arow0 + ks + hi8, arow0 + ks + 16 + hi8);
        v16h a1 = make_a_f16(arow1 + ks + hi8, arow1 + ks + 16 + hi8);
#pragma unroll
        for (int n = 0; n < 4; ++n) {
            v16h b = make_a_f16(wrow + (size_t)n * 16 * D_MOD + ks, wrow + (size_t)n * 16 * D_MOD + ks + 16);
            acc[0][n] = wmma_f16(a0, b, acc[0][n]);
            acc[1][n] = wmma_f16(a1, b, acc[1][n]);
        }
    }

#pragma unroll
    for (int n = 0; n < 4; ++n) {
        const int   j  = cn + n * 16 + l15;
        const float bj = bo[j];
#pragma unroll
        for (int i = 0; i < 2; ++i) {
#pragma unroll
            for (int r = 0; r < 8; ++r) {
                const int nr = tm * 32 + i * 16 + r + hi8;
                out[(size_t)nr * D_MOD + j] = acc[i][n][r] + bj;
            }
        }
    }
}

// ---------------------------------------------------------------------------
// Launch
// ---------------------------------------------------------------------------
extern "C" void kernel_launch(void* const* d_in, const int* in_sizes, int n_in,
                              void* d_out, int out_size, void* d_ws, size_t ws_size,
                              hipStream_t stream) {
    (void)in_sizes; (void)n_in; (void)out_size; (void)ws_size;

    const float* query = (const float*)d_in[0];
    const float* key_  = (const float*)d_in[1];
    const float* value = (const float*)d_in[2];
    const float* Wq    = (const float*)d_in[3];
    const float* bq    = (const float*)d_in[4];
    const float* Wk    = (const float*)d_in[5];
    const float* bk    = (const float*)d_in[6];
    const float* Wv    = (const float*)d_in[7];
    const float* bv    = (const float*)d_in[8];
    const float* Wo    = (const float*)d_in[9];
    const float* bo    = (const float*)d_in[10];
    float* out = (float*)d_out;

    char* ws = (char*)d_ws;
    _Float16* W16   = (_Float16*)ws;                              //  2 MB
    _Float16* q16   = (_Float16*)(ws + (size_t)2  * 1024 * 1024); // 16 MB
    _Float16* k16   = (_Float16*)(ws + (size_t)18 * 1024 * 1024); // 16 MB
    _Float16* vT16  = (_Float16*)(ws + (size_t)34 * 1024 * 1024); // 16 MB
    _Float16* att16 = (_Float16*)(ws + (size_t)50 * 1024 * 1024); // 16 MB

    const dim3 blk(32, 8);                       // 8 wave32 waves per block

    cvt_weights<<<1024, 256, 0, stream>>>(Wq, Wk, Wv, Wo, W16);

    // projections: 512 row-tiles(32) x 8 col-tiles(64) = 4096 waves -> 512 blocks
    proj_kernel<<<512, blk, 0, stream>>>(query, W16,              bq, q16,  0);
    proj_kernel<<<512, blk, 0, stream>>>(key_,  W16 + 262144,     bk, k16,  0);
    proj_kernel<<<512, blk, 0, stream>>>(value, W16 + 2 * 262144, bv, vT16, 1);

    // attention: 128 (b,h) x 8 q-tile-groups = 1024 blocks
    attn_kernel<<<1024, blk, 0, stream>>>(q16, k16, vT16, att16);

    // output projection
    oproj_kernel<<<512, blk, 0, stream>>>(att16, W16 + 3 * 262144, bo, out);
}
